// VectorQuantizer_86827058856705
// MI455X (gfx1250) — matrix operand order repaired and checked by reading, not compiled, not run on hardware
//
#include <hip/hip_runtime.h>

// ---------------------------------------------------------------------------
// VQ-VAE vector quantizer for MI455X (gfx1250, wave32, WMMA).
//   latents  : [32,4096,64] f32  (N=131072 rows, D=64)
//   codebook : [1024,64] f32     (K=1024)
//   out      : [N,64] f32 (q = codebook[argmin dist])  +  1 f32 vq_loss
// argmin(||x-c||^2) == argmax(x.c - 0.5||c||^2), scored on the bf16 WMMA
// pipe; candidate index packed into the low 10 mantissa bits of the score
// so the running argmax is a single v_max_num_f32 per 16x16 C register.
// Gather + loss epilogue in exact fp32 from the original codebook.
// ---------------------------------------------------------------------------

typedef __bf16 bf16_t;
typedef __attribute__((ext_vector_type(16))) __bf16 v16bf;
typedef __attribute__((ext_vector_type(8)))  __bf16 v8bf;
typedef __attribute__((ext_vector_type(8)))  float  v8f;

#define KC          1024          // codebook entries
#define DD          64            // embedding dim
#define NTOT        (32 * 4096)   // rows
#define CB_STRIDE   72            // LDS row pad: 144B = 36 dwords -> conflict-free B reads
#define WAVES       8
#define MT          2             // row-tiles (of 16 rows) per wave
#define ROWS_PER_WG (WAVES * MT * 16)    // 256
#define NBLOCKS     (NTOT / ROWS_PER_WG) // 512
#define LOSS_SCALE  (1.25f / (float)(NTOT * DD))
#define IDX_MASK    0x3FFu        // 10 bits for 1024 entries

// ---- prep: codebook f32 -> packed bf16 [1024][64] + half squared norms ----
__global__ void vq_prep(const float* __restrict__ cb,
                        bf16_t* __restrict__ cb_bf,
                        float* __restrict__ hn) {
  int e = blockIdx.x * blockDim.x + threadIdx.x;
  if (e >= KC) return;
  const float* r = cb + e * DD;
  float s = 0.f;
#pragma unroll
  for (int d = 0; d < DD; ++d) {
    float v = r[d];
    s += v * v;
    cb_bf[e * DD + d] = (bf16_t)v;
  }
  hn[e] = 0.5f * s;
}

// ---- main: WMMA scores + packed argmax + gather + partial loss ----
__global__ __launch_bounds__(256, 1)
void vq_main(const float* __restrict__ x,
             const float* __restrict__ cb,
             const bf16_t* __restrict__ cb_bf,
             const float* __restrict__ hn,
             float* __restrict__ out,
             float* __restrict__ partials) {
  __shared__ __align__(16) bf16_t sh_cb[KC * CB_STRIDE];  // 144 KB
  __shared__ __align__(16) float  sh_hn[KC];              // 4 KB
  __shared__ int   sh_idx[ROWS_PER_WG];                   // 1 KB
  __shared__ float sh_ps[WAVES];

  const int tid  = threadIdx.x;
  const int wave = tid >> 5;
  const int lane = tid & 31;
  const int ml   = lane & 15;   // column / row-in-tile index
  const int hi   = lane >> 4;   // half-wave select

  // --- async-copy bf16 codebook (packed [1024][64]) into padded LDS rows ---
#pragma unroll 1
  for (int c = tid; c < KC * 8; c += 256) {
    int row = c >> 3, ch = c & 7;
    unsigned lds_off = (unsigned)(size_t)(&sh_cb[row * CB_STRIDE]) + (unsigned)(ch * 16);
    unsigned goff    = (unsigned)(row * 128 + ch * 16);
    asm volatile("global_load_async_to_lds_b128 %0, %1, %2"
                 :: "v"(lds_off), "v"(goff), "s"(cb_bf) : "memory");
  }
  {
    unsigned lds_off = (unsigned)(size_t)(&sh_hn[0]) + (unsigned)(tid * 16);
    unsigned goff    = (unsigned)(tid * 16);
    asm volatile("global_load_async_to_lds_b128 %0, %1, %2"
                 :: "v"(lds_off), "v"(goff), "s"(hn) : "memory");
  }
  asm volatile("s_wait_asynccnt 0" ::: "memory");
  __syncthreads();

  // --- build A fragments: 16x32 bf16 per (row-tile, K-chunk) ---
  // lane l (ml,hi): elems 0..7  = x[row][kc + hi*8 + e]
  //                 elems 8..15 = x[row][kc + 16 + hi*8 + e]
  const int wgBase = blockIdx.x * ROWS_PER_WG;
  v16bf afrag[MT][2];
#pragma unroll
  for (int m = 0; m < MT; ++m) {
    int row = wgBase + (wave * MT + m) * 16 + ml;
    const float* xr = x + (size_t)row * DD;
#pragma unroll
    for (int k = 0; k < 2; ++k) {
      const float* xp = xr + k * 32 + hi * 8;
      v16bf a;
#pragma unroll
      for (int e = 0; e < 8; ++e) a[e] = (bf16_t)xp[e];
#pragma unroll
      for (int e = 0; e < 8; ++e) a[8 + e] = (bf16_t)xp[16 + e];
      afrag[m][k] = a;
    }
  }

  // running best: score with candidate index packed in low 10 mantissa bits
  float best[MT][8];
#pragma unroll
  for (int m = 0; m < MT; ++m)
#pragma unroll
    for (int v = 0; v < 8; ++v) best[m][v] = -3.4028235e38f;

  // --- sweep 64 code tiles of 16 entries each ---
#pragma unroll 2
  for (int t = 0; t < KC / 16; ++t) {
    const unsigned entry = (unsigned)(t * 16 + ml);
    // B 32x16 bf16 fragment: lane l holds column ml, K = hi*8 + 0..7, 16 + hi*8 + 0..7
    const bf16_t* bp = sh_cb + (size_t)entry * CB_STRIDE + hi * 8;
    v8bf p0 = *(const v8bf*)(bp);
    v8bf p1 = *(const v8bf*)(bp + 16);
    v8bf p2 = *(const v8bf*)(bp + 32);
    v8bf p3 = *(const v8bf*)(bp + 48);
    v16bf b0 = __builtin_shufflevector(p0, p1, 0,1,2,3,4,5,6,7,8,9,10,11,12,13,14,15);
    v16bf b1 = __builtin_shufflevector(p2, p3, 0,1,2,3,4,5,6,7,8,9,10,11,12,13,14,15);
    float hnl = sh_hn[t * 16 + ml];

#pragma unroll
    for (int m = 0; m < MT; ++m) {
      v8f acc = {0.f,0.f,0.f,0.f,0.f,0.f,0.f,0.f};
      acc = __builtin_amdgcn_wmma_f32_16x16x32_bf16(false, afrag[m][0], false, b0,
                                                    (short)0, acc, false, false);
      acc = __builtin_amdgcn_wmma_f32_16x16x32_bf16(false, afrag[m][1], false, b1,
                                                    (short)0, acc, false, false);
      // C layout: VGPR v, lane l -> row (v + 8*hi), col ml
#pragma unroll
      for (int v = 0; v < 8; ++v) {
        float sc = acc[v] - hnl;                    // maximize x.c - 0.5||c||^2
        unsigned pb = (__builtin_bit_cast(unsigned, sc) & ~IDX_MASK) | entry;
        best[m][v] = fmaxf(best[m][v], __builtin_bit_cast(float, pb));
      }
    }
  }

  // --- argmax across the 16 columns (lanes of same half-wave) ---
#pragma unroll
  for (int m = 0; m < MT; ++m) {
#pragma unroll
    for (int v = 0; v < 8; ++v) {
      float b = best[m][v];
#pragma unroll
      for (int off = 8; off >= 1; off >>= 1)
        b = fmaxf(b, __shfl_xor(b, off, 32));
      if (ml == 0)
        sh_idx[(wave * MT + m) * 16 + v + 8 * hi] =
            (int)(__builtin_bit_cast(unsigned, b) & IDX_MASK);
    }
  }
  __syncthreads();

  // --- gather q (exact fp32 codebook), write output, accumulate loss ---
  float psum = 0.f;
#pragma unroll 1
  for (int m = 0; m < MT; ++m) {
    int rbase = wgBase + (wave * MT + m) * 16;
#pragma unroll 1
    for (int r = 0; r < 16; ++r) {
      int idx = sh_idx[(wave * MT + m) * 16 + r];
      int row = rbase + r;
      float2 q  = *(const float2*)(cb + (size_t)idx * DD + lane * 2);
      float2 xv = *(const float2*)(x  + (size_t)row * DD + lane * 2);
      *(float2*)(out + (size_t)row * DD + lane * 2) = q;
      float dx = q.x - xv.x, dy = q.y - xv.y;
      psum += dx * dx + dy * dy;
    }
  }
#pragma unroll
  for (int off = 16; off >= 1; off >>= 1) psum += __shfl_xor(psum, off, 32);
  if (lane == 0) sh_ps[wave] = psum;
  __syncthreads();
  if (tid == 0) {
    float s = 0.f;
#pragma unroll
    for (int w = 0; w < WAVES; ++w) s += sh_ps[w];
    partials[blockIdx.x] = s;   // deterministic per-block partial
  }
}

// ---- deterministic fixed-order loss finalize ----
__global__ void vq_finalize(const float* __restrict__ partials,
                            float* __restrict__ loss) {
  if (threadIdx.x == 0 && blockIdx.x == 0) {
    float s = 0.f;
    for (int i = 0; i < NBLOCKS; ++i) s += partials[i];
    loss[0] = s * LOSS_SCALE;
  }
}

extern "C" void kernel_launch(void* const* d_in, const int* in_sizes, int n_in,
                              void* d_out, int out_size, void* d_ws, size_t ws_size,
                              hipStream_t stream) {
  (void)in_sizes; (void)n_in; (void)out_size; (void)ws_size;
  const float* x  = (const float*)d_in[0];   // latents  [N,64] f32
  const float* cb = (const float*)d_in[1];   // codebook [1024,64] f32
  float* out = (float*)d_out;                // [N,64] q  then 1 loss scalar

  // workspace layout: bf16 codebook | half-norms | per-block partials
  bf16_t* cb_bf    = (bf16_t*)d_ws;                                   // 128 KB
  float*  hn       = (float*)((char*)d_ws + KC * DD * 2);             // 4 KB
  float*  partials = (float*)((char*)d_ws + KC * DD * 2 + KC * 4);    // 2 KB
  float*  loss     = out + (size_t)NTOT * DD;

  vq_prep<<<(KC + 255) / 256, 256, 0, stream>>>(cb, cb_bf, hn);
  vq_main<<<NBLOCKS, 256, 0, stream>>>(x, cb, cb_bf, hn, out, partials);
  vq_finalize<<<1, 32, 0, stream>>>(partials, loss);
}
